// PromptConditionedMultiScaleAdapter_29549374997128
// MI455X (gfx1250) — matrix operand order repaired and checked by reading, not compiled, not run on hardware
//
// PromptConditionedMultiScaleAdapter — MI455X (gfx1250, wave32, WMMA) implementation.
//
// All matmuls run through v_wmma_f32_16x16x32_f16 (f16 operands staged in LDS
// with f32->f16 conversion on load, f32 accumulate). GEMM uses a 128x64 block
// tile: 8 wave32s, each owning a 32x32 sub-tile (2 A-frags x 2 B-frags = 4 WMMA
// per K-step), double-buffered LDS (30 KB). Attention (QK^T, softmax, PV) is
// batched per (batch*head) in chunks of 32 to bound workspace. Non-GEMM glue
// (rmsnorm/softmax/conv-dw/pos-embed/gate/top-k) uses wave32 shfl reductions.
//
// Input leaf order assumes jax pytree flatten (dicts sorted by key):
//  0 curves, 1 feat0, 2 feat1, 3 feat2,
//  4 gate_prm_b, 5 gate_prm_w, 6 gate_tok_w, 7 geom_b1, 8 geom_b2,
//  9 geom_norm, 10 geom_w1, 11 geom_w2,
//  12..23 refine[0]{down_b,down_w,gate_b,gate_w,mlp_norm,mlp_scale,sa_norm,sa_scale,wk,wo,wq,wv},
//  24 s0.dw, 25..36 s0.layer0, 37..48 s0.layer1, 49 s0.out_b, 50 s0.out_w, 51 s0.pe_gauss, 52 s0.pw,
//  53 s1.dw, 54..65 s1.layer0, 66..77 s1.layer1, 78 s1.out_b, 79 s1.out_w, 80 s1.pe_gauss, 81 s1.pw,
//  82..93 s2.layer0, 94..105 s2.layer1, 106 s2.out_b, 107 s2.out_w, 108 s2.pe_gauss,
//  109 score_prm_w, 110 score_tok_w, 111 type_emb.

#include <hip/hip_runtime.h>
#include <math.h>

typedef __attribute__((ext_vector_type(16))) _Float16 v16h;
typedef __attribute__((ext_vector_type(8)))  _Float16 v8h;
typedef __attribute__((ext_vector_type(8)))  float    v8f;

#define TPB 256
#define NWAVE (TPB / 32)

enum { EPI_NONE = 0, EPI_BIAS = 1, EPI_BIAS_SILU = 2, EPI_RES_SCALED = 3, EPI_BIAS_RES_SCALED = 4 };

// ---------------------------------------------------------------------------
// wave32 reductions
// ---------------------------------------------------------------------------
__device__ inline float block_reduce_add(float v, float* part)
{
#pragma unroll
  for (int o = 16; o > 0; o >>= 1) v += __shfl_xor(v, o, 32);
  __syncthreads();                       // protect 'part' from a previous use
  if ((threadIdx.x & 31) == 0) part[threadIdx.x >> 5] = v;
  __syncthreads();
  float s = 0.f;
#pragma unroll
  for (int i = 0; i < NWAVE; i++) s += part[i];
  return s;
}

__device__ inline float block_reduce_max(float v, float* part)
{
#pragma unroll
  for (int o = 16; o > 0; o >>= 1) v = fmaxf(v, __shfl_xor(v, o, 32));
  __syncthreads();
  if ((threadIdx.x & 31) == 0) part[threadIdx.x >> 5] = v;
  __syncthreads();
  float s = part[0];
#pragma unroll
  for (int i = 1; i < NWAVE; i++) s = fmaxf(s, part[i]);
  return s;
}

// ---------------------------------------------------------------------------
// WMMA GEMM: C[M,N] = epilogue( A[M,K] @ (TRANSB ? B[N,K]^T : B[K,N]) )
// f32 in/out, f16 compute, f32 accumulate. Requires M%128==0, N%64==0, K%32==0.
// Batched over grid.z: global z = z0+blockIdx.z, decomposed (bb,hh)=divmod(z,nh),
// with independent strides per local-z / bb / hh for A, B, C.
// Block tile 128x64: wave (wm 0..3, wn 0..1) computes rows wm*32..+31,
// cols wn*32..+31 -> acc[2][2] of 16x16 f32.
// ---------------------------------------------------------------------------
template <bool TRANSB>
__global__ __launch_bounds__(TPB) void gemm_wmma(
    const float* __restrict__ A, const float* __restrict__ B, float* __restrict__ C,
    int M, int N, int K, int lda, int ldb, int ldc,
    int z0, int nh,
    long sAz, long sAb, long sAh,
    long sBz, long sBb, long sBh,
    long sCz, long sCb, long sCh,
    const float* __restrict__ bias, const float* __restrict__ residual,
    const float* __restrict__ alpha_ptr, float cscale, int epi)
{
  __shared__ _Float16 As[2][128][40];  // [m][k], padded rows (80B, 16B aligned)
  __shared__ _Float16 Bs[2][64][40];   // [n][k]

  const int tid  = threadIdx.x;
  const int lane = tid & 31;
  const int wave = tid >> 5;
  const int wm   = wave >> 1;  // 0..3 -> 32-row strip
  const int wn   = wave & 1;   // 0..1 -> 32-col strip
  const int m0   = blockIdx.y * 128;
  const int n0   = blockIdx.x * 64;

  {
    const int zg = z0 + (int)blockIdx.z;
    const int bb = zg / nh;
    const int hh = zg - bb * nh;
    A += (long)blockIdx.z * sAz + (long)bb * sAb + (long)hh * sAh;
    B += (long)blockIdx.z * sBz + (long)bb * sBb + (long)hh * sBh;
    C += (long)blockIdx.z * sCz + (long)bb * sCb + (long)hh * sCh;
  }

  v8f acc[2][2] = {};

  const int ar  = tid >> 3;        // 0..31 (row group for A / trans-B tile loads)
  const int ac4 = (tid & 7) * 4;   // 0..28 (k offset, 4 floats)
  const int bn4 = (tid & 15) * 4;  // 0..60 (n offset for normal-B loads)
  const int bk  = tid >> 4;        // 0..15 (k row for normal-B loads)

  auto load_tiles = [&](int nb, int kk) {
#pragma unroll
    for (int rr = ar; rr < 128; rr += 32) {
      const float4 f = *(const float4*)(A + (long)(m0 + rr) * lda + kk + ac4);
      _Float16* d = &As[nb][rr][ac4];
      d[0] = (_Float16)f.x; d[1] = (_Float16)f.y; d[2] = (_Float16)f.z; d[3] = (_Float16)f.w;
    }
    if (TRANSB) {
#pragma unroll
      for (int rr = ar; rr < 64; rr += 32) {
        const float4 f = *(const float4*)(B + (long)(n0 + rr) * ldb + kk + ac4);
        _Float16* d = &Bs[nb][rr][ac4];
        d[0] = (_Float16)f.x; d[1] = (_Float16)f.y; d[2] = (_Float16)f.z; d[3] = (_Float16)f.w;
      }
    } else {
#pragma unroll
      for (int kr = bk; kr < 32; kr += 16) {
        const float4 f = *(const float4*)(B + (long)(kk + kr) * ldb + n0 + bn4);
        Bs[nb][bn4 + 0][kr] = (_Float16)f.x;
        Bs[nb][bn4 + 1][kr] = (_Float16)f.y;
        Bs[nb][bn4 + 2][kr] = (_Float16)f.z;
        Bs[nb][bn4 + 3][kr] = (_Float16)f.w;
      }
    }
  };

  load_tiles(0, 0);
  __syncthreads();

  // Fragment addressing per CDNA5 ISA layouts:
  //  A 16x32 f16: lane&15 = M row; halves 0-7 -> K = 8*(lane>>4)+0..7, halves 8-15 -> +16.
  //  B 32x16 f16: lane&15 = N col; halves 0-7 -> K = 8*(lane>>4)+0..7, halves 8-15 -> +16  (Bs stored [n][k]).
  const int ml0 = wm * 32 + (lane & 15);
  const int nl0 = wn * 32 + (lane & 15);
  const int akg = (lane >> 4) * 8;
  const int bkg = (lane >> 4) * 8;

  auto fragA = [&](int nb, int row) {
    const v8h lo = *(const v8h*)&As[nb][row][akg];
    const v8h hi = *(const v8h*)&As[nb][row][akg + 16];
    v16h r;
#pragma unroll
    for (int i = 0; i < 8; i++) { r[i] = lo[i]; r[8 + i] = hi[i]; }
    return r;
  };
  auto fragB = [&](int nb, int col) {
    const v8h lo = *(const v8h*)&Bs[nb][col][bkg];
    const v8h hi = *(const v8h*)&Bs[nb][col][bkg + 16];
    v16h r;
#pragma unroll
    for (int i = 0; i < 8; i++) { r[i] = lo[i]; r[8 + i] = hi[i]; }
    return r;
  };

  for (int kk = 0; kk < K; kk += 32) {
    const int nb = (kk >> 5) & 1;
    if (kk + 32 < K) {
      if (kk + 64 < K && tid == 0) {
        __builtin_prefetch(A + (long)m0 * lda + kk + 64, 0, 1);
        if (TRANSB) __builtin_prefetch(B + (long)n0 * ldb + kk + 64, 0, 1);
        else        __builtin_prefetch(B + (long)(kk + 64) * ldb + n0, 0, 1);
      }
      load_tiles(nb ^ 1, kk + 32);
    }

    v16h a[2], b[2];
    a[0] = fragA(nb, ml0);
    a[1] = fragA(nb, ml0 + 16);
    b[0] = fragB(nb, nl0);
    b[1] = fragB(nb, nl0 + 16);
#pragma unroll
    for (int mi = 0; mi < 2; mi++)
#pragma unroll
      for (int ti = 0; ti < 2; ti++)
        acc[mi][ti] = __builtin_amdgcn_wmma_f32_16x16x32_f16(
            false, a[mi], false, b[ti], (short)0, acc[mi][ti], false, false);
    __syncthreads();
  }

  // Epilogue. C 16x16 f32 layout: lane&15 = N; vgpr r -> M = r + 8*(lane>>4).
  float al = 1.0f;
  if ((epi == EPI_RES_SCALED || epi == EPI_BIAS_RES_SCALED) && alpha_ptr) al = tanhf(alpha_ptr[0]);
  const int mbase = m0 + wm * 32 + ((lane >> 4) << 3);
  const int nc0   = n0 + wn * 32 + (lane & 15);
#pragma unroll
  for (int mi = 0; mi < 2; mi++) {
#pragma unroll
    for (int r = 0; r < 8; r++) {
      const int m = mbase + mi * 16 + r;
#pragma unroll
      for (int ti = 0; ti < 2; ti++) {
        const int n = nc0 + ti * 16;
        float v2 = acc[mi][ti][r] * cscale;
        if (epi == EPI_BIAS || epi == EPI_BIAS_SILU || epi == EPI_BIAS_RES_SCALED) v2 += bias[n];
        if (epi == EPI_BIAS_SILU) v2 = v2 / (1.0f + __expf(-v2));
        if (epi == EPI_RES_SCALED || epi == EPI_BIAS_RES_SCALED) v2 = residual[(long)m * ldc + n] + al * v2;
        C[(long)m * ldc + n] = v2;
      }
    }
  }
}

// ---------------------------------------------------------------------------
// Elementwise / reduction kernels
// ---------------------------------------------------------------------------
__global__ __launch_bounds__(TPB) void rmsnorm_kernel(const float* __restrict__ x,
                                                      const float* __restrict__ scale,
                                                      float* __restrict__ y, int E)
{
  const long row = blockIdx.x;
  const float* xr = x + row * (long)E;
  float* yr = y + row * (long)E;
  const int tid = threadIdx.x;
  __shared__ float part[NWAVE];
  float s = 0.f;
  for (int i = tid; i < E; i += TPB) { const float v = xr[i]; s += v * v; }
  const float tot = block_reduce_add(s, part);
  const float inv = rsqrtf(tot / (float)E + 1e-5f);
  for (int i = tid; i < E; i += TPB) yr[i] = xr[i] * inv * scale[i];
}

__global__ __launch_bounds__(TPB) void softmax_rows(float* __restrict__ p, int S)
{
  const long row = blockIdx.x;
  float* pr = p + row * (long)S;
  const int tid = threadIdx.x;
  __shared__ float part[NWAVE];
  __shared__ float buf[512];
  float m = -INFINITY;
  for (int i = tid; i < S; i += TPB) { const float v = pr[i]; buf[i] = v; m = fmaxf(m, v); }
  const float mx = block_reduce_max(m, part);
  float s = 0.f;
  for (int i = tid; i < S; i += TPB) { const float e = __expf(buf[i] - mx); buf[i] = e; s += e; }
  const float tot = block_reduce_add(s, part);
  const float inv = 1.0f / tot;
  for (int i = tid; i < S; i += TPB) pr[i] = buf[i] * inv;
}

__global__ __launch_bounds__(TPB) void dwconv_kernel(const float* __restrict__ in,
                                                     const float* __restrict__ w,
                                                     float* __restrict__ out,
                                                     int C, int Hin, int df)
{
  const int Ho = Hin / df;
  const long i = (long)blockIdx.x * TPB + threadIdx.x;
  const long total = 8L * C * Ho * Ho;
  if (i >= total) return;
  const int wo = (int)(i % Ho);
  long t = i / Ho;
  const int ho = (int)(t % Ho); t /= Ho;
  const int c = (int)(t % C);
  const int b = (int)(t / C);
  const float* ip = in + (((long)b * C + c) * Hin + (long)ho * df) * Hin + (long)wo * df;
  const float* wp = w + (long)c * df * df;
  float s = 0.f;
  for (int ky = 0; ky < df; ky++)
    for (int kx = 0; kx < df; kx++)
      s += ip[(long)ky * Hin + kx] * wp[ky * df + kx];
  out[i] = s;
}

__global__ __launch_bounds__(TPB) void transpose_cs(const float* __restrict__ in,
                                                    float* __restrict__ out, int C, int S)
{
  const long i = (long)blockIdx.x * TPB + threadIdx.x;
  const long total = 8L * C * S;
  if (i >= total) return;
  const int s = (int)(i % S);
  long t = i / S;
  const int c = (int)(t % C);
  const int b = (int)(t / C);
  out[((long)b * S + s) * C + c] = in[i];
}

__global__ __launch_bounds__(TPB) void posembed_add(float* __restrict__ h,
                                                    const float* __restrict__ gauss)
{
  const int bs = blockIdx.x;        // b*256 + s, 16x16 grid
  const int s = bs & 255;
  const int y = s >> 4, x = s & 15;
  const float gx = ((x + 0.5f) / 16.f) * 2.f - 1.f;
  const float gy = ((y + 0.5f) / 16.f) * 2.f - 1.f;
  float* hr = h + (long)bs * 1024;
  for (int d = threadIdx.x; d < 1024; d += TPB) {
    const int d2 = d & 511;
    const float pr = (gx * gauss[d2] + gy * gauss[512 + d2]) * 6.28318530717958647692f;
    hr[d] += (d < 512) ? __sinf(pr) : __cosf(pr);
  }
}

__global__ __launch_bounds__(TPB) void gate_apply(float* __restrict__ h,
                                                  const float* __restrict__ g,
                                                  const float* __restrict__ pg,
                                                  int D, int S)
{
  const long i = (long)blockIdx.x * TPB + threadIdx.x;
  const long total = 8L * S * D;
  if (i >= total) return;
  const int d = (int)(i % D);
  const long bs = i / D;
  const int b = (int)(bs / S);
  const float z = g[i] + pg[(long)b * D + d];
  h[i] *= 1.0f / (1.0f + __expf(-z));
}

__global__ __launch_bounds__(TPB) void rowdot(const float* __restrict__ t,
                                              const float* __restrict__ ps,
                                              float* __restrict__ sc,
                                              int D, int S, float scale)
{
  const int row = blockIdx.x;
  const int b = row / S;
  const float* tr = t + (long)row * D;
  const float* pr = ps + (long)b * D;
  const int tid = threadIdx.x;
  __shared__ float part[NWAVE];
  float s = 0.f;
  for (int i = tid; i < D; i += TPB) s += tr[i] * pr[i];
  const float tot = block_reduce_add(s, part);
  if (tid == 0) sc[row] = tot * scale;
}

__global__ void geom_features(const float* __restrict__ curves, float* __restrict__ gf)
{
  const int b = threadIdx.x;
  if (b >= 8) return;
  const float* c = curves + b * 8;
  float xs[4], ys[4];
  for (int i = 0; i < 4; i++) { xs[i] = c[2 * i]; ys[i] = c[2 * i + 1]; }
  float xmin = xs[0], xmax = xs[0], ymin = ys[0], ymax = ys[0];
  for (int i = 1; i < 4; i++) {
    xmin = fminf(xmin, xs[i]); xmax = fmaxf(xmax, xs[i]);
    ymin = fminf(ymin, ys[i]); ymax = fmaxf(ymax, ys[i]);
  }
  const float cx = 0.5f * (xmin + xmax), cy = 0.5f * (ymin + ymax);
  const float w = fmaxf(xmax - xmin, 1e-6f), h = fmaxf(ymax - ymin, 1e-6f);
  float* g = gf + b * 20;
  for (int i = 0; i < 8; i++) g[i] = c[i];
  g[8] = xmin; g[9] = ymin; g[10] = xmax; g[11] = ymax;
  g[12] = cx; g[13] = cy; g[14] = w; g[15] = h;
  g[16] = w * h; g[17] = w / h;
  g[18] = xs[3] - xs[0]; g[19] = ys[3] - ys[0];
}

// Tiny dense layer for the 8-row prompt pipeline: Y[b,:] = act(X[b,:]@W + bias + addrow)
__global__ __launch_bounds__(TPB) void small_mm(const float* __restrict__ X,
                                                const float* __restrict__ W,
                                                const float* __restrict__ bias,
                                                const float* __restrict__ addrow,
                                                float* __restrict__ Y,
                                                int K, int N, int act)
{
  const int b = blockIdx.x;
  for (int j = threadIdx.x; j < N; j += TPB) {
    float s = bias ? bias[j] : 0.f;
    for (int k2 = 0; k2 < K; k2++) s += X[(long)b * K + k2] * W[(long)k2 * N + j];
    if (addrow) s += addrow[j];
    if (act) s = s / (1.0f + __expf(-s));
    Y[(long)b * N + j] = s;
  }
}

// Bitonic top-k (S==256): sort (score desc, idx asc) in LDS then gather rows.
__global__ __launch_bounds__(TPB) void topk_gather(const float* __restrict__ sc,
                                                   const float* __restrict__ h,
                                                   float* __restrict__ mem,
                                                   int S, int D, int kkeep, int out_off)
{
  __shared__ float vals[256];
  __shared__ int   idxs[256];
  const int b = blockIdx.x;
  const int tid = threadIdx.x;
  vals[tid] = sc[(long)b * S + tid];
  idxs[tid] = tid;
  for (int ksz = 2; ksz <= 256; ksz <<= 1) {
    for (int j = ksz >> 1; j > 0; j >>= 1) {
      __syncthreads();
      const int ixj = tid ^ j;
      if (ixj > tid) {
        const bool up = ((tid & ksz) != 0);
        const float vi = vals[tid], vj = vals[ixj];
        const int ai = idxs[tid], aj = idxs[ixj];
        const bool before = (vi > vj) || (vi == vj && ai < aj);
        if (up ? before : !before) {
          vals[tid] = vj; vals[ixj] = vi;
          idxs[tid] = aj; idxs[ixj] = ai;
        }
      }
    }
  }
  __syncthreads();
  for (int e = tid; e < kkeep * D; e += TPB) {
    const int jj = e / D, dd = e - jj * D;
    mem[((long)b * 512 + out_off + jj) * D + dd] = h[((long)b * S + idxs[jj]) * D + dd];
  }
}

// ---------------------------------------------------------------------------
// Host-side orchestration
// ---------------------------------------------------------------------------
static void launch_gemm(hipStream_t st, bool transB,
                        const float* A, const float* B, float* C,
                        int M, int N, int K, int lda, int ldb, int ldc,
                        int zc, int z0, int nh,
                        long sAz, long sAb, long sAh,
                        long sBz, long sBb, long sBh,
                        long sCz, long sCb, long sCh,
                        const float* bias, const float* res, const float* alpha,
                        float cs, int epi)
{
  dim3 g(N / 64, M / 128, zc), blk(TPB, 1, 1);
  if (transB)
    gemm_wmma<true><<<g, blk, 0, st>>>(A, B, C, M, N, K, lda, ldb, ldc, z0, nh,
                                       sAz, sAb, sAh, sBz, sBb, sBh, sCz, sCb, sCh,
                                       bias, res, alpha, cs, epi);
  else
    gemm_wmma<false><<<g, blk, 0, st>>>(A, B, C, M, N, K, lda, ldb, ldc, z0, nh,
                                        sAz, sAb, sAh, sBz, sBb, sBh, sCz, sCb, sCh,
                                        bias, res, alpha, cs, epi);
}

static void gemm_w(hipStream_t st, const float* A, const float* B, float* C,
                   int M, int N, int K,
                   const float* bias, const float* res, const float* alpha,
                   float cs, int epi)
{
  launch_gemm(st, false, A, B, C, M, N, K, K, N, N,
              1, 0, 1, 0, 0, 0, 0, 0, 0, 0, 0, 0,
              bias, res, alpha, cs, epi);
}

struct LayerW {
  const float *down_b, *down_w, *gate_b, *gate_w, *mlp_norm, *mlp_scale,
              *sa_norm, *sa_scale, *wk, *wo, *wq, *wv;
};
static inline LayerW layer_at(void* const* d_in, int i)
{
  auto f = [&](int j) { return (const float*)d_in[i + j]; };
  LayerW L;
  L.down_b = f(0);  L.down_w = f(1);  L.gate_b = f(2);   L.gate_w = f(3);
  L.mlp_norm = f(4); L.mlp_scale = f(5); L.sa_norm = f(6); L.sa_scale = f(7);
  L.wk = f(8); L.wo = f(9); L.wq = f(10); L.wv = f(11);
  return L;
}

// Pre-norm transformer layer, in-place on x (final MLP output optionally to outC).
static void run_tlayer(hipStream_t st, const LayerW& L,
                       float* x, float* xn, float* q, float* kb, float* vb,
                       float* att, float* attno, float* mlph, float* outC,
                       int Bb, int S, int E, int NH)
{
  const int R = Bb * S;
  const int HD = E / NH;   // always 64 here
  rmsnorm_kernel<<<R, TPB, 0, st>>>(x, L.sa_norm, xn, E);
  gemm_w(st, xn, L.wq, q,  R, E, E, nullptr, nullptr, nullptr, 1.f, EPI_NONE);
  gemm_w(st, xn, L.wk, kb, R, E, E, nullptr, nullptr, nullptr, 1.f, EPI_NONE);
  gemm_w(st, xn, L.wv, vb, R, E, E, nullptr, nullptr, nullptr, 1.f, EPI_NONE);

  const int ZT = Bb * NH;
  const float iscale = 0.125f;  // 1/sqrt(64)
  for (int z0 = 0; z0 < ZT; z0 += 32) {
    const int zc = (ZT - z0 < 32) ? (ZT - z0) : 32;
    // scores[z] = Q_z @ K_z^T * iscale  -> att (chunk-local)
    launch_gemm(st, true, q, kb, att, S, S, HD, E, E, S,
                zc, z0, NH,
                0, (long)S * E, HD,
                0, (long)S * E, HD,
                (long)S * S, 0, 0,
                nullptr, nullptr, nullptr, iscale, EPI_NONE);
    softmax_rows<<<zc * S, TPB, 0, st>>>(att, S);
    // attno[z] = P_z @ V_z
    launch_gemm(st, false, att, vb, attno, S, HD, S, S, E, E,
                zc, z0, NH,
                (long)S * S, 0, 0,
                0, (long)S * E, HD,
                0, (long)S * E, HD,
                nullptr, nullptr, nullptr, 1.f, EPI_NONE);
  }
  // x = x + tanh(sa_scale) * (attno @ wo)
  gemm_w(st, attno, L.wo, x, R, E, E, nullptr, x, L.sa_scale, 1.f, EPI_RES_SCALED);
  // MLP
  rmsnorm_kernel<<<R, TPB, 0, st>>>(x, L.mlp_norm, xn, E);
  gemm_w(st, xn, L.gate_w, mlph, R, 4 * E, E, L.gate_b, nullptr, nullptr, 1.f, EPI_BIAS_SILU);
  float* dst = outC ? outC : x;
  gemm_w(st, mlph, L.down_w, dst, R, E, 4 * E, L.down_b, x, L.mlp_scale, 1.f, EPI_BIAS_RES_SCALED);
}

extern "C" void kernel_launch(void* const* d_in, const int* in_sizes, int n_in,
                              void* d_out, int out_size, void* d_ws, size_t ws_size,
                              hipStream_t stream)
{
  (void)in_sizes; (void)n_in; (void)out_size; (void)ws_size;
  auto in = [&](int i) { return (const float*)d_in[i]; };

  // ---- input leaves ----
  const float* curves = in(0);
  const float* feat[3] = { in(1), in(2), in(3) };
  const float* gate_prm_b = in(4);
  const float* gate_prm_w = in(5);
  const float* gate_tok_w = in(6);
  const float* geom_b1 = in(7);
  const float* geom_b2 = in(8);
  const float* geom_norm = in(9);
  const float* geom_w1 = in(10);
  const float* geom_w2 = in(11);
  const LayerW refineL = layer_at(d_in, 12);
  const int dw_i[3]   = { 24, 53, -1 };
  const int l0_i[3]   = { 25, 54, 82 };
  const int l1_i[3]   = { 37, 66, 94 };
  const int outb_i[3] = { 49, 78, 106 };
  const int outw_i[3] = { 50, 79, 107 };
  const int pe_i[3]   = { 51, 80, 108 };
  const int pw_i[3]   = { 52, 81, -1 };
  const float* score_prm_w = in(109);
  const float* score_tok_w = in(110);
  const float* type_emb = in(111);

  // ---- workspace carve-up ----
  char* base = (char*)d_ws;
  size_t off = 0;
  auto alloc = [&](size_t nfloat) {
    float* p = (float*)(base + off);
    off += ((nfloat * sizeof(float)) + 255) & ~(size_t)255;
    return p;
  };
  float* x     = alloc(2048L * 768);       // per-scale tokens [B*256, 768]
  float* xn    = alloc(4096L * 1024);      // rmsnorm scratch (refine-sized)
  float* q     = alloc(4096L * 1024);
  float* kb    = alloc(4096L * 1024);
  float* vb    = alloc(4096L * 1024);
  float* att   = alloc(32L * 512 * 512);   // chunked attention scores
  float* attno = alloc(4096L * 1024);
  float* mlph  = alloc(4096L * 4096);
  float* dwout = alloc(8L * 768 * 256);
  float* pwout = alloc(8L * 768 * 256);
  float* tok[3] = { alloc(2048L * 1024), alloc(2048L * 1024), alloc(2048L * 1024) };
  float* gbuf  = alloc(2048L * 1024);
  float* sc    = alloc(3L * 8 * 256);
  float* gf    = alloc(8L * 20);
  float* pv1   = alloc(8L * 1024);
  float* pvb   = alloc(8L * 1024);
  float* prompt= alloc(8L * 1024);
  float* pg    = alloc(8L * 1024);
  float* ps    = alloc(8L * 1024);
  float* mem   = alloc(8L * 512 * 1024);

  // ---- prompt pipeline (tiny: 8 rows) ----
  geom_features<<<1, 32, 0, stream>>>(curves, gf);
  small_mm<<<8, TPB, 0, stream>>>(gf, geom_w1, geom_b1, nullptr, pv1, 20, 1024, 1);
  small_mm<<<8, TPB, 0, stream>>>(pv1, geom_w2, geom_b2, type_emb /*row 0*/, pvb, 1024, 1024, 0);
  rmsnorm_kernel<<<8, TPB, 0, stream>>>(pvb, geom_norm, prompt, 1024);
  small_mm<<<8, TPB, 0, stream>>>(prompt, gate_prm_w, gate_prm_b, nullptr, pg, 1024, 1024, 0);
  small_mm<<<8, TPB, 0, stream>>>(prompt, score_prm_w, nullptr, nullptr, ps, 1024, 1024, 0);

  // ---- per-scale token pipelines ----
  const int Hin_s[3] = { 64, 32, 16 };
  const int df_s[3]  = { 4, 2, 1 };
  const long tot_tok = 8L * 768 * 256;
  for (int si = 0; si < 3; si++) {
    const float* src;
    if (df_s[si] > 1) {
      dwconv_kernel<<<(int)((tot_tok + TPB - 1) / TPB), TPB, 0, stream>>>(
          feat[si], in(dw_i[si]), dwout, 768, Hin_s[si], df_s[si]);
      // pointwise 1x1 conv as GEMM: pwout[b] = pw[768,768] @ dwout[b][768,256]
      launch_gemm(stream, false, in(pw_i[si]), dwout, pwout,
                  768, 256, 768, 768, 256, 256,
                  8, 0, 1,
                  0, 0, 0,
                  0, 768L * 256, 0,
                  0, 768L * 256, 0,
                  nullptr, nullptr, nullptr, 1.f, EPI_NONE);
      src = pwout;
    } else {
      src = feat[si];
    }
    transpose_cs<<<(int)((tot_tok + TPB - 1) / TPB), TPB, 0, stream>>>(src, x, 768, 256);

    run_tlayer(stream, layer_at(d_in, l0_i[si]), x, xn, q, kb, vb, att, attno, mlph,
               nullptr, 8, 256, 768, 12);
    run_tlayer(stream, layer_at(d_in, l1_i[si]), x, xn, q, kb, vb, att, attno, mlph,
               nullptr, 8, 256, 768, 12);

    // project to decoder dim, add fourier pos-embed, prompt-gate, score
    gemm_w(stream, x, in(outw_i[si]), tok[si], 2048, 1024, 768,
           in(outb_i[si]), nullptr, nullptr, 1.f, EPI_BIAS);
    posembed_add<<<2048, TPB, 0, stream>>>(tok[si], in(pe_i[si]));
    gemm_w(stream, tok[si], gate_tok_w, gbuf, 2048, 1024, 1024,
           nullptr, nullptr, nullptr, 1.f, EPI_NONE);
    gate_apply<<<(int)((2048L * 1024) / TPB), TPB, 0, stream>>>(tok[si], gbuf, pg, 1024, 256);
    gemm_w(stream, tok[si], score_tok_w, gbuf, 2048, 1024, 1024,
           nullptr, nullptr, nullptr, 1.f, EPI_NONE);
    rowdot<<<2048, TPB, 0, stream>>>(gbuf, ps, sc + (long)si * 2048, 1024, 256, 0.03125f);
  }

  // ---- static top-k allocation from allocate_topk([256,256,256], 512, 16) ----
  const int ks_[3] = { 171, 171, 170 };
  const int ofs[3] = { 0, 171, 342 };
  for (int si = 0; si < 3; si++)
    topk_gather<<<8, TPB, 0, stream>>>(sc + (long)si * 2048, tok[si], mem,
                                       256, 1024, ks_[si], ofs[si]);

  // ---- refine layer on [8,512,1024], final MLP writes d_out ----
  run_tlayer(stream, refineL, mem, xn, q, kb, vb, att, attno, mlph,
             (float*)d_out, 8, 512, 1024, 16);
}
